// SwinTransformerBlock_84679575208279
// MI455X (gfx1250) — matrix operand order repaired and checked by reading, not compiled, not run on hardware
//
#include <hip/hip_runtime.h>
#include <hip/hip_bf16.h>

typedef __attribute__((ext_vector_type(16))) _Float16 v16h;
typedef __attribute__((ext_vector_type(8)))  _Float16 v8h;
typedef __attribute__((ext_vector_type(8)))  float    v8f;
typedef __attribute__((ext_vector_type(4)))  int      v4i;

#define M_TOK 200704L   // 64 * 3136 tokens (= 4096 windows * 49)
#define C_DIM 128
#define N_WIN 49
#define EPS   1e-5f

// ---- CDNA5 async global->LDS path (guarded; falls back to plain loads) ----
#if defined(__has_builtin)
# if __has_builtin(__builtin_amdgcn_global_load_async_to_lds_b128)
#  define USE_ASYNC 1
# endif
#endif
#ifndef USE_ASYNC
# define USE_ASYNC 0
#endif

// builtin signature (from hipcc diagnostic): param0 = AS(1) v4i*, then LDS ptr, offset, cpol
#define GLOBAL_V4(p) ((__attribute__((address_space(1))) v4i*)(p))
#define LDS_V4(p)    ((__attribute__((address_space(3))) v4i*)(p))

__device__ __forceinline__ void wait_asynccnt0() {
#if USE_ASYNC
# if __has_builtin(__builtin_amdgcn_s_wait_asynccnt)
    __builtin_amdgcn_s_wait_asynccnt(0);
# else
    asm volatile("s_wait_asynccnt 0" ::: "memory");
# endif
#endif
}

// ---------------------------------------------------------------------------
// fragment helpers (contiguous k = 16*half + s, same order for A and B)
// ---------------------------------------------------------------------------
__device__ __forceinline__ v16h cat8(v8h lo, v8h hi) {
    return __builtin_shufflevector(lo, hi, 0,1,2,3,4,5,6,7,8,9,10,11,12,13,14,15);
}

// A-matrix 16x32 f16: lanes 0-15 rows; elems = K(half*16 .. +15) contiguous
__device__ __forceinline__ v16h load_a_frag(const _Float16* __restrict__ p, int lda,
                                            int ml, int half) {
    const _Float16* pa = p + (long)ml * lda + half * 16;
    v8h lo = *(const v8h*)pa;
    v8h hi = *(const v8h*)(pa + 8);
    return cat8(lo, hi);
}

// B-matrix 32x16 f16: lane = column n, elems = K(half*16 .. +15) contiguous
__device__ __forceinline__ v16h load_b_frag(const _Float16* __restrict__ p, int ldb,
                                            int nl, int half) {
    return *(const v16h*)(p + (long)nl * ldb + half * 16);
}

__device__ __forceinline__ v8f wmma_f16(v16h a, v16h b, v8f c) {
    return __builtin_amdgcn_wmma_f32_16x16x32_f16(false, a, false, b, (short)0, c,
                                                  false, false);
}

// windowed row index -> original token index (window reverse + unshift)
__device__ __forceinline__ long win_to_tok(long m) {
    long w = m / 49; int i = (int)(m % 49);
    long b = w >> 6; int wi = (int)(w & 63);
    int wr = wi >> 3, wc = wi & 7;
    int hs = wr * 7 + i / 7;
    int ws_ = wc * 7 + i % 7;
    int h = hs + 3; if (h >= 56) h -= 56;
    int ww = ws_ + 3; if (ww >= 56) ww -= 56;
    return b * 3136 + h * 56 + ww;
}

__device__ __forceinline__ int region_id(int p) {   // shifted-image region (0,1,2)
    return (p < 49) ? 0 : ((p < 53) ? 1 : 2);
}

// ---------------------------------------------------------------------------
// fp32 -> f16 weight conversion
// ---------------------------------------------------------------------------
__global__ __launch_bounds__(256) void cvt_f16(const float* __restrict__ in,
                                               _Float16* __restrict__ out, int n) {
    int i = blockIdx.x * 256 + threadIdx.x;
    if (i < n) out[i] = (_Float16)in[i];
}

// ---------------------------------------------------------------------------
// LayerNorm1 + cyclic shift + window partition (fp32 -> f16 windowed order)
// ---------------------------------------------------------------------------
__global__ __launch_bounds__(256)
void ln1_shift_win(const float* __restrict__ x, const float* __restrict__ g,
                   const float* __restrict__ bt, _Float16* __restrict__ out) {
    long t = (long)blockIdx.x * 8 + (threadIdx.x >> 5);
    int lane = threadIdx.x & 31;
    const float4 xv = *(const float4*)(x + t * C_DIM + lane * 4);
    float s  = xv.x + xv.y + xv.z + xv.w;
    float s2 = xv.x*xv.x + xv.y*xv.y + xv.z*xv.z + xv.w*xv.w;
    for (int o = 16; o > 0; o >>= 1) { s += __shfl_xor(s, o, 32); s2 += __shfl_xor(s2, o, 32); }
    float mu = s * (1.f / C_DIM);
    float rs = rsqrtf(s2 * (1.f / C_DIM) - mu * mu + EPS);
    long b = t / 3136; int sp = (int)(t % 3136);
    int h = sp / 56, w = sp % 56;
    int hp = h + 53; if (hp >= 56) hp -= 56;   // (h-3) mod 56
    int wp = w + 53; if (wp >= 56) wp -= 56;
    int win = (hp / 7) * 8 + wp / 7;
    int ii  = (hp % 7) * 7 + wp % 7;
    _Float16* o4 = out + ((b * 64 + win) * 49L + ii) * C_DIM + lane * 4;
    const float4 gg = *(const float4*)(g + lane * 4);
    const float4 bb = *(const float4*)(bt + lane * 4);
    o4[0] = (_Float16)((xv.x - mu) * rs * gg.x + bb.x);
    o4[1] = (_Float16)((xv.y - mu) * rs * gg.y + bb.y);
    o4[2] = (_Float16)((xv.z - mu) * rs * gg.z + bb.z);
    o4[3] = (_Float16)((xv.w - mu) * rs * gg.w + bb.w);
}

// LayerNorm2: token order in (fp32) -> token order out (f16)
__global__ __launch_bounds__(256)
void ln2_tok(const float* __restrict__ x, const float* __restrict__ g,
             const float* __restrict__ bt, _Float16* __restrict__ out) {
    long t = (long)blockIdx.x * 8 + (threadIdx.x >> 5);
    int lane = threadIdx.x & 31;
    const float4 xv = *(const float4*)(x + t * C_DIM + lane * 4);
    float s  = xv.x + xv.y + xv.z + xv.w;
    float s2 = xv.x*xv.x + xv.y*xv.y + xv.z*xv.z + xv.w*xv.w;
    for (int o = 16; o > 0; o >>= 1) { s += __shfl_xor(s, o, 32); s2 += __shfl_xor(s2, o, 32); }
    float mu = s * (1.f / C_DIM);
    float rs = rsqrtf(s2 * (1.f / C_DIM) - mu * mu + EPS);
    _Float16* o4 = out + t * C_DIM + lane * 4;
    const float4 gg = *(const float4*)(g + lane * 4);
    const float4 bb = *(const float4*)(bt + lane * 4);
    o4[0] = (_Float16)((xv.x - mu) * rs * gg.x + bb.x);
    o4[1] = (_Float16)((xv.y - mu) * rs * gg.y + bb.y);
    o4[2] = (_Float16)((xv.z - mu) * rs * gg.z + bb.z);
    o4[3] = (_Float16)((xv.w - mu) * rs * gg.w + bb.w);
}

// ---------------------------------------------------------------------------
// Generic WMMA GEMM: Y[M,Nf] = A[M,K] @ W[Nf,K]^T  (+ epilogue)
// wave tile: 32(M) x 64(N); 8 waves/block; K-step 32
// ---------------------------------------------------------------------------
template<int EPI>
__global__ __launch_bounds__(256)
void gemm_wmma(const _Float16* __restrict__ A, const _Float16* __restrict__ W,
               const float* __restrict__ bias, int Nf, int K,
               _Float16* __restrict__ outH, float* __restrict__ outF,
               const float* __restrict__ resid) {
    const int wave = threadIdx.x >> 5;
    const int lane = threadIdx.x & 31;
    const int ml   = lane & 15;
    const int half = lane >> 4;
    const long m0  = ((long)blockIdx.x * 8 + wave) * 32;
    const int  n0  = blockIdx.y * 64;

    v8f acc[2][4] = {};

    for (int k0 = 0; k0 < K; k0 += 32) {
        if (k0 + 32 < K)
            __builtin_prefetch(A + (m0 + ml) * (long)K + k0 + 32, 0, 1);
        v16h a0 = load_a_frag(A + m0 * K + k0,          K, ml, half);
        v16h a1 = load_a_frag(A + (m0 + 16) * K + k0,   K, ml, half);
        v16h b0 = load_b_frag(W + (long)(n0     ) * K + k0, K, ml, half);
        v16h b1 = load_b_frag(W + (long)(n0 + 16) * K + k0, K, ml, half);
        v16h b2 = load_b_frag(W + (long)(n0 + 32) * K + k0, K, ml, half);
        v16h b3 = load_b_frag(W + (long)(n0 + 48) * K + k0, K, ml, half);
        acc[0][0] = wmma_f16(a0, b0, acc[0][0]);
        acc[0][1] = wmma_f16(a0, b1, acc[0][1]);
        acc[0][2] = wmma_f16(a0, b2, acc[0][2]);
        acc[0][3] = wmma_f16(a0, b3, acc[0][3]);
        acc[1][0] = wmma_f16(a1, b0, acc[1][0]);
        acc[1][1] = wmma_f16(a1, b1, acc[1][1]);
        acc[1][2] = wmma_f16(a1, b2, acc[1][2]);
        acc[1][3] = wmma_f16(a1, b3, acc[1][3]);
    }

    const int nl = lane & 15;
    const int mh = lane >> 4;
    for (int sT = 0; sT < 2; sT++) {
        for (int tT = 0; tT < 4; tT++) {
            const int col = n0 + tT * 16 + nl;
            const float bc = bias[col];
            const long mrb = m0 + sT * 16 + mh * 8;
            for (int i = 0; i < 8; i++) {
                const long m = mrb + i;
                float v = acc[sT][tT][i] + bc;
                if (EPI == 0) {
                    outH[m * Nf + col] = (_Float16)v;
                } else if (EPI == 1) {
                    long t = win_to_tok(m);
                    outF[t * C_DIM + col] = v + resid[t * C_DIM + col];
                } else if (EPI == 2) {
                    v = 0.5f * v * (1.f + erff(v * 0.70710678118654752f));
                    outH[m * Nf + col] = (_Float16)v;
                } else {
                    outF[m * C_DIM + col] = v + resid[m * C_DIM + col];
                }
            }
        }
    }
}

// ---------------------------------------------------------------------------
// Window attention: one block (128 thr, 4 waves) per (window, head)
// QKV staged into LDS via GLOBAL_LOAD_ASYNC_TO_LDS_B128 (ASYNCcnt), while the
// bias+mask table is precomputed into sS; sS is then fed as the WMMA C-matrix
// (no divergent epilogue around v_wmma; EXEC stays all-ones).
// ---------------------------------------------------------------------------
__global__ __launch_bounds__(128)
void attn_kernel(const _Float16* __restrict__ qkv, const float* __restrict__ rpb,
                 _Float16* __restrict__ outH) {
    const long win = blockIdx.x >> 2;          // 0..4095 global window
    const int head = blockIdx.x & 3;

    __shared__ __align__(16) _Float16 sQ[64][32];
    __shared__ __align__(16) _Float16 sK[64][32];
    __shared__ __align__(16) _Float16 sV[64][32];
    __shared__ __align__(16) _Float16 sVt[32][64];   // transposed V
    __shared__ __align__(16) float    sS[64][64];    // bias+mask, then scores, then exp
    __shared__ __align__(16) _Float16 sP[64][64];

    const int tid = threadIdx.x;

#if USE_ASYNC
    {   // async DMA rows of Q/K/V into LDS: 3 mats * 49 rows * 4 x b128
        const _Float16* gbase = qkv + (win * 49) * 384 + head * 32;
        for (int ci = tid; ci < 3 * 49 * 4; ci += 128) {
            int mat = ci / 196, rem = ci % 196;
            int r = rem >> 2, c8 = (rem & 3) * 8;
            const _Float16* gsrc = gbase + (long)r * 384 + mat * 128 + c8;
            _Float16* ldst = (mat == 0) ? &sQ[r][c8] : (mat == 1) ? &sK[r][c8] : &sV[r][c8];
            __builtin_amdgcn_global_load_async_to_lds_b128(GLOBAL_V4(gsrc), LDS_V4(ldst),
                                                           0, 0);
        }
    }
#endif

    // zero the padded rows (49..63) of Q/K/V and padded keys of Vt (overlaps DMA)
    for (int idx = tid; idx < 15 * 32; idx += 128) {
        int r = 49 + (idx >> 5), c = idx & 31;
        sQ[r][c] = (_Float16)0.f; sK[r][c] = (_Float16)0.f; sV[r][c] = (_Float16)0.f;
        sVt[c][r] = (_Float16)0.f;
    }

    // prefill sS with rel-pos bias + shift mask (C-matrix of the score WMMA);
    // padded rows/cols get -1e5 so softmax zeroes them. Overlaps the async DMA.
    {
        const int wi = (int)(win & 63);
        const int wr = wi >> 3, wc = wi & 7;
        for (int idx = tid; idx < 64 * 64; idx += 128) {
            int row = idx >> 6, col = idx & 63;
            float v = -1e5f;
            if (row < N_WIN && col < N_WIN) {
                int yi = row / 7, xi = row % 7, yj = col / 7, xj = col % 7;
                float bias = rpb[((yi - yj + 6) * 13 + (xi - xj + 6)) * 4 + head];
                int cnti = 3 * region_id(wr * 7 + yi) + region_id(wc * 7 + xi);
                int cntj = 3 * region_id(wr * 7 + yj) + region_id(wc * 7 + xj);
                v = bias + ((cnti != cntj) ? -100.f : 0.f);
            }
            sS[row][col] = v;
        }
    }

#if USE_ASYNC
    wait_asynccnt0();
#else
    for (int idx = tid; idx < N_WIN * 32; idx += 128) {
        int r = idx >> 5, c = idx & 31;
        long base = (win * 49 + r) * 384 + head * 32 + c;
        sQ[r][c] = qkv[base]; sK[r][c] = qkv[base + 128]; sV[r][c] = qkv[base + 256];
    }
#endif
    __syncthreads();

    // fold 1/sqrt(HD) into Q; build transposed V for the P@V B-fragments
    for (int idx = tid; idx < N_WIN * 32; idx += 128) {
        int r = idx >> 5, c = idx & 31;
        sQ[r][c] = (_Float16)((float)sQ[r][c] * 0.17677669529663687f);
        sVt[c][r] = sV[r][c];
    }
    __syncthreads();

    const int wave = tid >> 5, lane = tid & 31;
    const int ml = lane & 15, half = lane >> 4;
    const int m0 = wave * 16;
    const int nl = ml, mh = half;

    // ---- S = (Q*scale) K^T + (bias+mask from C operand) ----
    v16h aq = load_a_frag(&sQ[m0][0], 32, ml, half);
    for (int nt = 0; nt < 4; nt++) {
        v16h bk = *(const v16h*)&sK[nt * 16 + ml][half * 16];
        v8f c;
        for (int i = 0; i < 8; i++) c[i] = sS[m0 + mh * 8 + i][nt * 16 + nl];
        c = wmma_f16(aq, bk, c);
        for (int i = 0; i < 8; i++) sS[m0 + mh * 8 + i][nt * 16 + nl] = c[i];
    }
    __syncthreads();

    // ---- softmax over keys, one row per thread ----
    if (tid < 64) {
        const int r = tid;
        if (r < N_WIN) {
            float mx = -1e30f;
            for (int c = 0; c < 64; c++) mx = fmaxf(mx, sS[r][c]);
            float sum = 0.f;
            for (int c = 0; c < 64; c++) { float e = __expf(sS[r][c] - mx); sum += e; sS[r][c] = e; }
            float inv = 1.f / sum;
            for (int c = 0; c < 64; c++) sP[r][c] = (_Float16)(sS[r][c] * inv);
        } else {
            for (int c = 0; c < 64; c++) sP[r][c] = (_Float16)0.f;
        }
    }
    __syncthreads();

    // ---- O = P @ V ----
    v8f oacc[2] = {};
    for (int ks = 0; ks < 2; ks++) {
        v16h ap = load_a_frag(&sP[m0][ks * 32], 64, ml, half);
        v16h bv0 = *(const v16h*)&sVt[ml     ][ks * 32 + half * 16];
        v16h bv1 = *(const v16h*)&sVt[16 + ml][ks * 32 + half * 16];
        oacc[0] = wmma_f16(ap, bv0, oacc[0]);
        oacc[1] = wmma_f16(ap, bv1, oacc[1]);
    }
    for (int nt = 0; nt < 2; nt++) {
        const int d = head * 32 + nt * 16 + nl;
        for (int i = 0; i < 8; i++) {
            const int row = m0 + mh * 8 + i;
            if (row < N_WIN)
                outH[(win * 49 + row) * C_DIM + d] = (_Float16)oacc[nt][i];
        }
    }
}

// ---------------------------------------------------------------------------
// launcher
// ---------------------------------------------------------------------------
extern "C" void kernel_launch(void* const* d_in, const int* in_sizes, int n_in,
                              void* d_out, int out_size, void* d_ws, size_t ws_size,
                              hipStream_t stream) {
    const float* x       = (const float*)d_in[0];
    const float* n1g     = (const float*)d_in[1];
    const float* n1b     = (const float*)d_in[2];
    const float* qkv_w   = (const float*)d_in[3];
    const float* qkv_b   = (const float*)d_in[4];
    const float* rpb     = (const float*)d_in[5];
    const float* proj_w  = (const float*)d_in[6];
    const float* proj_b  = (const float*)d_in[7];
    const float* n2g     = (const float*)d_in[8];
    const float* n2b     = (const float*)d_in[9];
    const float* fc1_w   = (const float*)d_in[10];
    const float* fc1_b   = (const float*)d_in[11];
    const float* fc2_w   = (const float*)d_in[12];
    const float* fc2_b   = (const float*)d_in[13];
    float* out           = (float*)d_out;

    char* ws = (char*)d_ws;
    size_t off = 0;
    _Float16* wQKV = (_Float16*)(ws + off); off += 384 * 128 * sizeof(_Float16);
    _Float16* wPRJ = (_Float16*)(ws + off); off += 128 * 128 * sizeof(_Float16);
    _Float16* wFC1 = (_Float16*)(ws + off); off += 512 * 128 * sizeof(_Float16);
    _Float16* wFC2 = (_Float16*)(ws + off); off += 128 * 512 * sizeof(_Float16);
    off = (off + 255) & ~size_t(255);
    // region A: hwin -> attnout -> ln2out   (M x 128 f16)
    _Float16* regA = (_Float16*)(ws + off); off += M_TOK * 128 * sizeof(_Float16);
    // region B: qkv (M x 384 f16) -> fc1out (M x 512 f16)
    _Float16* regB = (_Float16*)(ws + off); off += M_TOK * 512 * sizeof(_Float16);
    // region C: X1 (M x 128 fp32)
    float* X1 = (float*)(ws + off); off += M_TOK * 128 * sizeof(float);

    cvt_f16<<<(384 * 128 + 255) / 256, 256, 0, stream>>>(qkv_w,  wQKV, 384 * 128);
    cvt_f16<<<(128 * 128 + 255) / 256, 256, 0, stream>>>(proj_w, wPRJ, 128 * 128);
    cvt_f16<<<(512 * 128 + 255) / 256, 256, 0, stream>>>(fc1_w,  wFC1, 512 * 128);
    cvt_f16<<<(128 * 512 + 255) / 256, 256, 0, stream>>>(fc2_w,  wFC2, 128 * 512);

    const int gmx = (int)(M_TOK / 32 / 8);   // 784
    ln1_shift_win<<<(int)(M_TOK / 8), 256, 0, stream>>>(x, n1g, n1b, regA);
    gemm_wmma<0><<<dim3(gmx, 384 / 64), 256, 0, stream>>>(regA, wQKV, qkv_b, 384, 128,
                                                          regB, nullptr, nullptr);
    attn_kernel<<<4096 * 4, 128, 0, stream>>>(regB, rpb, regA);
    gemm_wmma<1><<<dim3(gmx, 128 / 64), 256, 0, stream>>>(regA, wPRJ, proj_b, 128, 128,
                                                          nullptr, X1, x);
    ln2_tok<<<(int)(M_TOK / 8), 256, 0, stream>>>(X1, n2g, n2b, regA);
    gemm_wmma<2><<<dim3(gmx, 512 / 64), 256, 0, stream>>>(regA, wFC1, fc1_b, 512, 128,
                                                          regB, nullptr, nullptr);
    gemm_wmma<3><<<dim3(gmx, 128 / 64), 256, 0, stream>>>(regB, wFC2, fc2_b, 128, 512,
                                                          nullptr, out, X1);
    (void)in_sizes; (void)n_in; (void)out_size; (void)ws_size;
}
